// DynamicVQVAE_1795296330063
// MI455X (gfx1250) — matrix operand order repaired and checked by reading, not compiled, not run on hardware
//
#include <hip/hip_runtime.h>

typedef unsigned short u16;
typedef __attribute__((ext_vector_type(16))) __bf16 v16bf;
typedef __attribute__((ext_vector_type(8)))  float  v8f;

__device__ __forceinline__ u16 f2bf(float f) {
  union { float f; unsigned u; } v; v.f = f;
  unsigned r = v.u + 0x7fffu + ((v.u >> 16) & 1u);
  return (u16)(r >> 16);
}

// ---------------- layout conversion ----------------
__global__ void k_nchw_to_pixc(const float* __restrict__ x, u16* __restrict__ act,
                               int B, int C, int HW) {
  long long i = (long long)blockIdx.x * blockDim.x + threadIdx.x;
  long long total = (long long)B * C * HW;
  if (i >= total) return;
  int c = (int)(i % C);
  long long pix = i / C;
  int n = (int)(pix / HW);
  int p = (int)(pix % HW);
  act[i] = f2bf(x[((long long)n * C + c) * HW + p]);
}

// weights [Cout][Kreal] f32 -> [Npad][Kpad] bf16 (zero-padded)
__global__ void k_repack_w(const float* __restrict__ w, u16* __restrict__ out,
                           int Cout, int Kreal, int Kpad, int Npad) {
  long long i = (long long)blockIdx.x * blockDim.x + threadIdx.x;
  long long total = (long long)Npad * Kpad;
  if (i >= total) return;
  int k  = (int)(i % Kpad);
  int co = (int)(i / Kpad);
  u16 v = 0;
  if (co < Cout && k < Kreal) v = f2bf(w[(long long)co * Kreal + k]);
  out[i] = v;
}

// transposed-conv weights, torch layout [Cin][Cout][3][3] -> [Npad][Kpad] bf16
// (kernel-flip folded into the im2col relation, so weights copied unflipped)
__global__ void k_repack_wT(const float* __restrict__ w, u16* __restrict__ out,
                            int Cin, int Cout, int Kpad, int Npad) {
  long long i = (long long)blockIdx.x * blockDim.x + threadIdx.x;
  long long total = (long long)Npad * Kpad;
  if (i >= total) return;
  int k  = (int)(i % Kpad);
  int co = (int)(i / Kpad);
  u16 v = 0;
  if (co < Cout && k < Cin * 9) {
    int ci = k / 9, r = k % 9;
    v = f2bf(w[((long long)ci * Cout + co) * 9 + r]);
  }
  out[i] = v;
}

// ---------------- im2col (stride-2 conv, k=3, pad=1) ----------------
__global__ void k_im2col_s2(const u16* __restrict__ act, u16* __restrict__ col,
                            int Hin, int Win, int Cin, int Hout, int Wout,
                            int Kpad, long long pix0, long long rows) {
  long long i = (long long)blockIdx.x * blockDim.x + threadIdx.x;
  long long total = rows * Kpad;
  if (i >= total) return;
  int k = (int)(i % Kpad);
  long long pix = pix0 + i / Kpad;
  u16 v = 0;
  if (k < Cin * 9) {
    int ci = k / 9, r = k % 9, kh = r / 3, kw = r % 3;
    int HW = Hout * Wout;
    int n = (int)(pix / HW);
    int p = (int)(pix % HW);
    int oh = p / Wout, ow = p % Wout;
    int ih = oh * 2 + kh - 1, iw = ow * 2 + kw - 1;
    if (ih >= 0 && ih < Hin && iw >= 0 && iw < Win)
      v = act[(((long long)n * Hin + ih) * Win + iw) * Cin + ci];
  }
  col[i] = v;
}

// im2col for ConvTranspose2d(k=3,s=2,p=1,op=1): input index ih2 = oh+1-kh (even, in-range)
__global__ void k_im2col_t(const u16* __restrict__ act, u16* __restrict__ col,
                           int Hin, int Win, int Cin, int Hout, int Wout,
                           int Kpad, long long pix0, long long rows) {
  long long i = (long long)blockIdx.x * blockDim.x + threadIdx.x;
  long long total = rows * Kpad;
  if (i >= total) return;
  int k = (int)(i % Kpad);
  long long pix = pix0 + i / Kpad;
  u16 v = 0;
  if (k < Cin * 9) {
    int ci = k / 9, r = k % 9, kh = r / 3, kw = r % 3;
    int HW = Hout * Wout;
    int n = (int)(pix / HW);
    int p = (int)(pix % HW);
    int oh = p / Wout, ow = p % Wout;
    int ih2 = oh + 1 - kh, iw2 = ow + 1 - kw;
    if (ih2 >= 0 && iw2 >= 0 && !(ih2 & 1) && !(iw2 & 1)) {
      int ih = ih2 >> 1, iw = iw2 >> 1;
      if (ih < Hin && iw < Win)
        v = act[(((long long)n * Hin + ih) * Win + iw) * Cin + ci];
    }
  }
  col[i] = v;
}

// ---------------- WMMA GEMM: C[M,N] = A[M,Kpad](bf16) * W[N,Kpad](bf16)^T + bias ----------------
// Register-blocked 1 M-tile x NT N-tiles per wave; NT specialized at compile time so the
// K loop is branch-free (A fragment loaded once, reused across NT WMMAs).
union FragB { v16bf v; uint4 q[2]; };

template <int NT>
__device__ __forceinline__ void gemm_k_loop(const u16* __restrict__ aPtr,
                                            const u16* __restrict__ bPtr,
                                            size_t bStride, int Kpad, v8f* accs) {
  for (int k0 = 0; k0 < Kpad; k0 += 32) {
    FragB a;
    a.q[0] = *(const uint4*)(aPtr + k0);
    a.q[1] = *(const uint4*)(aPtr + k0 + 8);
    __builtin_prefetch(aPtr + k0 + 128, 0, 1);   // next A cachelines (global_prefetch_b8)
#pragma unroll
    for (int nt = 0; nt < NT; ++nt) {
      FragB b;
      b.q[0] = *(const uint4*)(bPtr + nt * bStride + k0);
      b.q[1] = *(const uint4*)(bPtr + nt * bStride + k0 + 8);
      accs[nt] = __builtin_amdgcn_wmma_f32_16x16x32_bf16(false, a.v, false, b.v,
                                                         (short)0, accs[nt], false, false);
    }
  }
}

__global__ void __launch_bounds__(128)
k_gemm_bf16(const u16* __restrict__ A, const u16* __restrict__ W,
            const float* __restrict__ bias, float* __restrict__ C,
            int Mrows, int Npad, int Kpad, int Nreal, int hasBias) {
  int wave = threadIdx.x >> 5;
  int lane = threadIdx.x & 31;
  int m0 = (blockIdx.x * 4 + wave) * 16;
  int n0 = blockIdx.y * 64;
  if (m0 >= Mrows) return;            // wave-uniform: EXEC stays all-ones for WMMA
  int nTiles = (Npad - n0) >> 4;      // wave-uniform, loop-invariant
  if (nTiles > 4) nTiles = 4;
  int half = lane & 15;
  // 16-bit A fragment as loaded here (contiguous k order): lanes0-15 hold K=0..15 of row M=lane, lanes16-31 K=16..31
  const u16* aPtr = A + (size_t)(m0 + half) * Kpad + ((lane < 16) ? 0 : 16);
  // 16-bit B fragment as loaded here (contiguous k order): lanes0-15 hold K=0..15 of col N=lane, lanes16-31 K=16..31
  const u16* bPtr = W + (size_t)(n0 + half) * Kpad + ((lane < 16) ? 0 : 16);
  size_t bStride = (size_t)16 * Kpad;
  v8f accs[4] = {v8f{}, v8f{}, v8f{}, v8f{}};
  switch (nTiles) {
    case 1:  gemm_k_loop<1>(aPtr, bPtr, bStride, Kpad, accs); break;
    case 2:  gemm_k_loop<2>(aPtr, bPtr, bStride, Kpad, accs); break;
    case 3:  gemm_k_loop<3>(aPtr, bPtr, bStride, Kpad, accs); break;
    default: gemm_k_loop<4>(aPtr, bPtr, bStride, Kpad, accs); break;
  }
  int rbase = m0 + ((lane < 16) ? 0 : 8);
  for (int nt = 0; nt < nTiles; ++nt) {
    int col = n0 + nt * 16 + half;
    float bv = 0.f;
    if (hasBias && col < Nreal) bv = bias[col];
#pragma unroll
    for (int r = 0; r < 8; ++r)
      C[(size_t)(rbase + r) * Npad + col] = accs[nt][r] + bv;
  }
}

// ---------------- batchnorm (training-mode, biased var) ----------------
__global__ void k_bn_stats(const float* __restrict__ conv, int M, int ldc,
                           const float* __restrict__ g, const float* __restrict__ beta,
                           float* __restrict__ scale, float* __restrict__ shift) {
  int c = blockIdx.x;
  float s = 0.f, ss = 0.f;
  for (int i = threadIdx.x; i < M; i += blockDim.x) {
    float v = conv[(size_t)i * ldc + c];
    s += v; ss += v * v;
  }
  __shared__ float rs[256], rss[256];
  rs[threadIdx.x] = s; rss[threadIdx.x] = ss;
  __syncthreads();
  for (int o = 128; o > 0; o >>= 1) {
    if (threadIdx.x < o) { rs[threadIdx.x] += rs[threadIdx.x + o]; rss[threadIdx.x] += rss[threadIdx.x + o]; }
    __syncthreads();
  }
  if (threadIdx.x == 0) {
    float inv = 1.f / (float)M;
    float mean = rs[0] * inv;
    float var = rss[0] * inv - mean * mean;
    float sc = g[c] * rsqrtf(var + 1e-5f);
    scale[c] = sc;
    shift[c] = beta[c] - mean * sc;
  }
}

__global__ void k_bn_lrelu(const float* __restrict__ conv, u16* __restrict__ act,
                           const float* __restrict__ scale, const float* __restrict__ shift,
                           int C, long long total) {
  long long i = (long long)blockIdx.x * blockDim.x + threadIdx.x;
  if (i >= total) return;
  int c = (int)(i % C);
  float v = conv[i] * scale[c] + shift[c];
  v = v > 0.f ? v : 0.2f * v;
  act[i] = f2bf(v);
}

__global__ void k_f2bf(const float* __restrict__ in, u16* __restrict__ out, long long n) {
  long long i = (long long)blockIdx.x * blockDim.x + threadIdx.x;
  if (i >= n) return;
  out[i] = f2bf(in[i]);
}

// ---------------- vector quantizer ----------------
__global__ void k_cbnorm(const float* __restrict__ cb, float* __restrict__ cbn, int K, int D) {
  int k = blockIdx.x * blockDim.x + threadIdx.x;
  if (k >= K) return;
  float s = 0.f;
  for (int d = 0; d < D; ++d) { float v = cb[(long long)k * D + d]; s += v * v; }
  cbn[k] = s;
}

__global__ void k_argmin(const float* __restrict__ dots, const float* __restrict__ cbn,
                         int* __restrict__ idx, int P, int Kc) {
  int p = blockIdx.x * blockDim.x + threadIdx.x;
  if (p >= P) return;
  float best = 3.4e38f; int bi = 0;
  for (int k = 0; k < Kc; ++k) {
    float d = cbn[k] - 2.f * dots[(long long)p * Kc + k];
    if (d < best) { best = d; bi = k; }
  }
  idx[p] = bi;
}

__global__ void k_gather_q(const float* __restrict__ cb, const int* __restrict__ idx,
                           u16* __restrict__ qb, int P, int D) {
  long long i = (long long)blockIdx.x * blockDim.x + threadIdx.x;
  if (i >= (long long)P * D) return;
  int d = (int)(i % D);
  int p = (int)(i / D);
  qb[i] = f2bf(cb[(long long)idx[p] * D + d]);
}

__global__ void k_zero(float* p) { if (threadIdx.x == 0 && blockIdx.x == 0) *p = 0.f; }

__global__ void k_vqloss(const float* __restrict__ ze, const float* __restrict__ cb,
                         const int* __restrict__ idx, float* __restrict__ out, int P, int D) {
  __shared__ float red[256];
  long long total = (long long)P * D;
  float s = 0.f;
  for (long long i = (long long)blockIdx.x * blockDim.x + threadIdx.x; i < total;
       i += (long long)gridDim.x * blockDim.x) {
    int d = (int)(i % D);
    int p = (int)(i / D);
    float diff = cb[(long long)idx[p] * D + d] - ze[i];
    s += diff * diff;
  }
  red[threadIdx.x] = s;
  __syncthreads();
  for (int o = 128; o > 0; o >>= 1) {
    if (threadIdx.x < o) red[threadIdx.x] += red[threadIdx.x + o];
    __syncthreads();
  }
  // vq_loss = q_loss + 0.25*e_loss; forward values identical -> 1.25 * mean((q - z_e)^2)
  if (threadIdx.x == 0) atomicAdd(out, red[0] * (1.25f / ((float)P * (float)D)));
}

// ---------------- outputs ----------------
__global__ void k_sigmoid_out(const float* __restrict__ conv, float* __restrict__ out,
                              int B, int H, int W) {
  long long i = (long long)blockIdx.x * blockDim.x + threadIdx.x;
  long long total = (long long)B * 3 * H * W;
  if (i >= total) return;
  int w = (int)(i % W);
  long long t = i / W;
  int h = (int)(t % H); t /= H;
  int c = (int)(t % 3);
  int n = (int)(t / 3);
  long long pix = ((long long)n * H + h) * W + w;
  float v = conv[pix * 16 + c];
  out[i] = 1.f / (1.f + __expf(-v));
}

__global__ void k_ze_out(const float* __restrict__ ze, float* __restrict__ out) {
  int i = blockIdx.x * blockDim.x + threadIdx.x;
  if (i >= 32 * 256 * 64) return;
  int w = i & 7; int t = i >> 3;
  int h = t & 7; t >>= 3;
  int c = t & 255; int n = t >> 8;
  out[i] = ze[((long long)n * 64 + h * 8 + w) * 256 + c];
}

__global__ void k_idx_out(const int* __restrict__ idx, float* __restrict__ out) {
  int i = blockIdx.x * blockDim.x + threadIdx.x;
  if (i >= 2048) return;
  out[i] = (float)idx[i];
}

// ---------------- host driver ----------------
static inline dim3 g1(long long n) { return dim3((unsigned)((n + 255) / 256)); }

extern "C" void kernel_launch(void* const* d_in, const int* in_sizes, int n_in,
                              void* d_out_v, int out_size, void* d_ws, size_t ws_size,
                              hipStream_t stream) {
  (void)in_sizes; (void)n_in; (void)out_size; (void)ws_size;
  const int B = 32;
  const float* x = (const float*)d_in[0];
  float* out = (float*)d_out_v;

  char* ws = (char*)d_ws;
  size_t off = 0;
  auto alloc = [&](size_t bytes) -> char* {
    char* p = ws + off;
    off += (bytes + 255) & ~(size_t)255;
    return p;
  };
  const size_t COL_CAP = 128ull << 20;
  u16*   colBuf  = (u16*)alloc(COL_CAP);
  float* convBuf = (float*)alloc(128ull << 20);
  u16*   actA    = (u16*)alloc(40ull << 20);
  u16*   actB    = (u16*)alloc(40ull << 20);
  u16*   wBuf    = (u16*)alloc(4ull << 20);
  float* scaleB  = (float*)alloc(512 * 4);
  float* shiftB  = (float*)alloc(512 * 4);
  float* zeBuf   = (float*)alloc(2048ull * 256 * 4);
  u16*   zeBf    = (u16*)alloc(2048ull * 256 * 2);
  u16*   cbBf    = (u16*)alloc(512ull * 256 * 2);
  float* cbN     = (float*)alloc(512 * 4);
  int*   idxBuf  = (int*)alloc(2048 * 4);
  u16*   qBf     = (u16*)alloc(2048ull * 256 * 2);

  dim3 blk(256);

  auto gemm_grid = [](long long rows, int Npad) {
    return dim3((unsigned)((rows / 16 + 3) / 4), (unsigned)((Npad + 63) / 64));
  };

  auto run_conv = [&](const u16* act, const float* w, const float* bias,
                      int Hin, int Win, int Cin, int Hout, int Wout, int Cout,
                      bool tr, float* convOut) {
    int Kreal = Cin * 9;
    int Kpad = (Kreal + 31) & ~31;
    int Npad = (Cout + 15) & ~15;
    long long M = (long long)B * Hout * Wout;
    long long wTot = (long long)Npad * Kpad;
    if (tr)
      k_repack_wT<<<g1(wTot), blk, 0, stream>>>(w, wBuf, Cin, Cout, Kpad, Npad);
    else
      k_repack_w<<<g1(wTot), blk, 0, stream>>>(w, wBuf, Cout, Kreal, Kpad, Npad);
    long long chunk = (long long)(COL_CAP / ((size_t)Kpad * 2)) & ~15LL;
    if (chunk > M) chunk = M;
    for (long long r0 = 0; r0 < M; r0 += chunk) {
      long long rows = (M - r0 < chunk) ? (M - r0) : chunk;
      long long cTot = rows * Kpad;
      if (tr)
        k_im2col_t<<<g1(cTot), blk, 0, stream>>>(act, colBuf, Hin, Win, Cin, Hout, Wout, Kpad, r0, rows);
      else
        k_im2col_s2<<<g1(cTot), blk, 0, stream>>>(act, colBuf, Hin, Win, Cin, Hout, Wout, Kpad, r0, rows);
      k_gemm_bf16<<<gemm_grid(rows, Npad), dim3(128), 0, stream>>>(colBuf, wBuf, bias,
          convOut + (size_t)r0 * Npad, (int)rows, Npad, Kpad, Cout, 1);
    }
  };

  // ---- encoder ----
  k_nchw_to_pixc<<<g1((long long)B * 3 * 256 * 256), blk, 0, stream>>>(x, actA, B, 3, 256 * 256);
  u16 *cur = actA, *nxt = actB;
  int H = 256, Cprev = 3;
  static const int encC[5] = {32, 64, 128, 256, 512};
  for (int i = 0; i < 5; ++i) {
    const float* w    = (const float*)d_in[1 + 4 * i];
    const float* b    = (const float*)d_in[2 + 4 * i];
    const float* g    = (const float*)d_in[3 + 4 * i];
    const float* beta = (const float*)d_in[4 + 4 * i];
    int Ho = H / 2, C = encC[i];
    run_conv(cur, w, b, H, H, Cprev, Ho, Ho, C, false, convBuf);
    long long M = (long long)B * Ho * Ho;
    k_bn_stats<<<dim3(C), blk, 0, stream>>>(convBuf, (int)M, C, g, beta, scaleB, shiftB);
    k_bn_lrelu<<<g1(M * C), blk, 0, stream>>>(convBuf, nxt, scaleB, shiftB, C, M * C);
    u16* t = cur; cur = nxt; nxt = t;
    H = Ho; Cprev = C;
  }

  // ---- proj 1x1 (512 -> 256), output straight into z_e buffer ----
  {
    const float* w = (const float*)d_in[21];
    const float* b = (const float*)d_in[22];
    k_repack_w<<<g1(256LL * 512), blk, 0, stream>>>(w, wBuf, 256, 512, 512, 256);
    k_gemm_bf16<<<gemm_grid(2048, 256), dim3(128), 0, stream>>>(cur, wBuf, b, zeBuf, 2048, 256, 512, 256, 1);
  }

  // ---- vector quantizer ----
  const float* cb = (const float*)d_in[23];
  k_f2bf<<<g1(2048LL * 256), blk, 0, stream>>>(zeBuf, zeBf, 2048LL * 256);
  k_f2bf<<<g1(512LL * 256), blk, 0, stream>>>(cb, cbBf, 512LL * 256);
  k_cbnorm<<<g1(512), blk, 0, stream>>>(cb, cbN, 512, 256);
  // dots = z_e . cb^T  (reuses convBuf)
  k_gemm_bf16<<<gemm_grid(2048, 512), dim3(128), 0, stream>>>(zeBf, cbBf, nullptr, convBuf, 2048, 512, 256, 512, 0);
  k_argmin<<<g1(2048), blk, 0, stream>>>(convBuf, cbN, idxBuf, 2048, 512);
  k_gather_q<<<g1(2048LL * 256), blk, 0, stream>>>(cb, idxBuf, qBf, 2048, 256);
  const long long LOSS_OFF = 6291456LL;
  k_zero<<<1, 1, 0, stream>>>(out + LOSS_OFF);
  k_vqloss<<<dim3(256), blk, 0, stream>>>(zeBuf, cb, idxBuf, out + LOSS_OFF, 2048, 256);

  // ---- dproj 1x1 (256 -> 512) on straight-through q ----
  {
    const float* w = (const float*)d_in[24];
    const float* b = (const float*)d_in[25];
    k_repack_w<<<g1(512LL * 256), blk, 0, stream>>>(w, wBuf, 512, 256, 256, 512);
    k_gemm_bf16<<<gemm_grid(2048, 512), dim3(128), 0, stream>>>(qBf, wBuf, b, convBuf, 2048, 512, 256, 512, 1);
  }
  k_f2bf<<<g1(2048LL * 512), blk, 0, stream>>>(convBuf, actA, 2048LL * 512);
  cur = actA; nxt = actB;

  // ---- decoder ----
  static const int decCin[5]  = {512, 256, 128, 64, 32};
  static const int decCout[5] = {256, 128, 64, 32, 3};
  H = 8;
  for (int i = 0; i < 5; ++i) {
    const float* w = (const float*)d_in[26 + 4 * i];
    const float* b = (const float*)d_in[27 + 4 * i];
    int Ho = H * 2;
    run_conv(cur, w, b, H, H, decCin[i], Ho, Ho, decCout[i], true, convBuf);
    long long M = (long long)B * Ho * Ho;
    if (i < 4) {
      const float* g    = (const float*)d_in[28 + 4 * i];
      const float* beta = (const float*)d_in[29 + 4 * i];
      int C = decCout[i];
      k_bn_stats<<<dim3(C), blk, 0, stream>>>(convBuf, (int)M, C, g, beta, scaleB, shiftB);
      k_bn_lrelu<<<g1(M * C), blk, 0, stream>>>(convBuf, nxt, scaleB, shiftB, C, M * C);
      u16* t = cur; cur = nxt; nxt = t;
    } else {
      k_sigmoid_out<<<g1((long long)B * 3 * 256 * 256), blk, 0, stream>>>(convBuf, out, B, 256, 256);
    }
    H = Ho;
  }

  // ---- remaining outputs: z_e (NCHW) and indices ----
  k_ze_out<<<g1(32LL * 256 * 64), blk, 0, stream>>>(zeBuf, out + (LOSS_OFF + 1));
  k_idx_out<<<g1(2048), blk, 0, stream>>>(idxBuf, out + (LOSS_OFF + 1 + 524288));
}
